// MultiHeadAttention_90555090469628
// MI455X (gfx1250) — matrix operand order repaired and checked by reading, not compiled, not run on hardware
//
#include <hip/hip_runtime.h>
#include <hip/hip_bf16.h>

typedef __bf16 bf16;
typedef __attribute__((ext_vector_type(16))) __bf16 v16bf;
typedef __attribute__((ext_vector_type(8)))  __bf16 v8bf;
typedef __attribute__((ext_vector_type(8)))  float   v8f;
typedef __attribute__((ext_vector_type(4)))  float   f32x4;   // native vector (not HIP_vector_type)

typedef __attribute__((ext_vector_type(4))) unsigned int u32x4;
typedef __attribute__((ext_vector_type(8))) int          i32x8;
typedef __attribute__((ext_vector_type(4))) int          i32x4;

// Global-address-space pointer typedefs: force global_load/global_store
// (flat_load couples LOADcnt with DScnt and was serializing the hot loops).
typedef const __attribute__((address_space(1))) float  gfloat;
typedef const __attribute__((address_space(1))) f32x4  gf32x4;
typedef const __attribute__((address_space(1))) bf16   gbf16;
typedef const __attribute__((address_space(1))) v8bf   gv8bf;
typedef __attribute__((address_space(1))) float        mgfloat;
typedef __attribute__((address_space(1))) bf16         mgbf16;
typedef __attribute__((address_space(1))) v8bf         mgv8bf;

#define BATCH 2
#define CCH   256
#define NH    8
#define HD    32
#define SEQ   4096

__device__ __forceinline__ v8f wmma_bf16(v16bf a, v16bf b, v8f c) {
    return __builtin_amdgcn_wmma_f32_16x16x32_bf16(false, a, false, b, (short)0, c, false, false);
}

__device__ __forceinline__ v16bf cat8(v8bf lo, v8bf hi) {
    v16bf r;
#pragma unroll
    for (int e = 0; e < 8; ++e) { r[e] = lo[e]; r[e + 8] = hi[e]; }
    return r;
}

// 16x32 A-operand chunk of a row-major fp32 matrix -> bf16, CDNA5 wave32 A layout:
// lane L: row = L%16; hal = L/16; elem e<8 -> K = hal*8+e ; e>=8 -> K = 16+hal*8+(e-8)
__device__ __forceinline__ v16bf load_w_tileA(const float* wr /* &W[row][c0 + hal*8] */) {
    gf32x4* p0 = (gf32x4*)wr;
    gf32x4* p1 = (gf32x4*)(wr + 16);
    f32x4 a0 = p0[0], a1 = p0[1];
    f32x4 a2 = p1[0], a3 = p1[1];
    v16bf a;
#pragma unroll
    for (int i = 0; i < 4; ++i) {
        a[i]      = (bf16)a0[i];
        a[4 + i]  = (bf16)a1[i];
        a[8 + i]  = (bf16)a2[i];
        a[12 + i] = (bf16)a3[i];
    }
    return a;
}

// ---------------------------------------------------------------------------
// Kernel 1: fused Q/K/V 1x1-conv projections.  x fp32 [B][C][S] -> q,k bf16
// [B][NH][S][HD], v transposed bf16 [B][NH][HD][S].
// ---------------------------------------------------------------------------
__global__ void __launch_bounds__(256) qkv_proj_kernel(
        const float* __restrict__ x,
        const float* __restrict__ wq, const float* __restrict__ bq,
        const float* __restrict__ wk, const float* __restrict__ bk,
        const float* __restrict__ wv, const float* __restrict__ bv,
        bf16* __restrict__ qo, bf16* __restrict__ ko, bf16* __restrict__ vt) {
    __shared__ __align__(16) bf16 xT[16][264];   // [s][c], padded: conflict-free b128 reads

    const int blk = blockIdx.x;
    const int b   = blk / (SEQ / 16);
    const int s0  = (blk % (SEQ / 16)) * 16;
    const int tid = threadIdx.x;

    // Stage x tile [256c x 16s] -> LDS transposed bf16 (coalesced global_load_b128).
    {
        const int c = tid;
        gf32x4* xv4 = (gf32x4*)(x + ((size_t)b * CCH + c) * SEQ + s0);
#pragma unroll
        for (int j4 = 0; j4 < 4; ++j4) {
            f32x4 v = xv4[j4];
            xT[j4 * 4 + 0][c] = (bf16)v[0];
            xT[j4 * 4 + 1][c] = (bf16)v[1];
            xT[j4 * 4 + 2][c] = (bf16)v[2];
            xT[j4 * 4 + 3][c] = (bf16)v[3];
        }
    }
    __syncthreads();

    const int w    = tid >> 5;
    const int lane = tid & 31;
    const int n    = lane & 15;
    const int hal  = lane >> 4;

    const float* Wm[3] = { wq, wk, wv };
    const float* Bm[3] = { bq, bk, bv };

    for (int mi = 0; mi < 2; ++mi) {
        const int cTile = (w * 2 + mi) * 16;
        for (int mat = 0; mat < 3; ++mat) {
            const float* W = Wm[mat];
            v8f acc = { 0.f, 0.f, 0.f, 0.f, 0.f, 0.f, 0.f, 0.f };
#pragma unroll
            for (int kc = 0; kc < 8; ++kc) {
                const int c0 = kc * 32;
                v16bf a = load_w_tileA(W + (size_t)(cTile + n) * CCH + c0 + hal * 8);
                const bf16* xp = &xT[n][c0 + hal * 8];
                v16bf bb = cat8(*reinterpret_cast<const v8bf*>(xp),
                                *reinterpret_cast<const v8bf*>(xp + 16));
                acc = wmma_bf16(a, bb, acc);
            }
            gfloat* bias = (gfloat*)Bm[mat];
            const int cbase = cTile + hal * 8;      // C-layout: c = cbase + r
            const int s = s0 + n;
#pragma unroll
            for (int r = 0; r < 8; ++r) acc[r] += bias[cbase + r];

            if (mat < 2) {
                bf16* dst = (mat == 0) ? qo : ko;
                const int h  = cbase / HD;
                const int d0 = cbase % HD;          // multiple of 8 -> 16B aligned
                v8bf ov;
#pragma unroll
                for (int r = 0; r < 8; ++r) ov[r] = (bf16)acc[r];
                *(mgv8bf*)(dst + (((size_t)(b * NH + h) * SEQ + s) * HD + d0)) = ov;
            } else {
#pragma unroll
                for (int r = 0; r < 8; ++r) {
                    const int c = cbase + r;
                    const int h = c / HD, d = c % HD;
                    ((mgbf16*)vt)[((size_t)(b * NH + h) * HD + d) * SEQ + s] = (bf16)acc[r];
                }
            }
        }
    }
}

// ---------------------------------------------------------------------------
// Kernel 2: flash attention.  One wave per 16-query tile, fully independent
// waves (no workgroup barriers in the hot loop).  hd=32 = one bf16 WMMA
// K-chunk: per 32 keys -> 2 score WMMAs + online softmax + 2 PV WMMAs.
// ---------------------------------------------------------------------------
__global__ void __launch_bounds__(256) flash_attn_kernel(
        const bf16* __restrict__ q, const bf16* __restrict__ k,
        const bf16* __restrict__ vt, bf16* __restrict__ aoT) {
    __shared__ __align__(16) bf16 P[8][16][40];  // per-wave P tile, stride 40 -> bank-clean

    const int blk  = blockIdx.x;
    const int nqb  = SEQ / 128;
    const int bh   = blk / nqb;
    const int qblk = blk % nqb;
    const int b    = bh / NH;
    const int h    = bh % NH;
    const int tid  = threadIdx.x;
    const int w    = tid >> 5;
    const int lane = tid & 31;
    const int n    = lane & 15;
    const int hal  = lane >> 4;
    const int q0   = qblk * 128 + w * 16;

    const float scale = 0.17677669529663687f;    // 1/sqrt(32)

    // Q tile in A layout, pre-scaled by softmax scale.
    v16bf qa;
    {
        const bf16* qp = q + (((size_t)(b * NH + h) * SEQ) + q0 + n) * HD;
        v8bf lo = *(gv8bf*)(qp + hal * 16);
        v8bf hi = *(gv8bf*)(qp + hal * 16 + 8);
#pragma unroll
        for (int e = 0; e < 8; ++e) {
            qa[e]     = (bf16)((float)lo[e] * scale);
            qa[e + 8] = (bf16)((float)hi[e] * scale);
        }
    }

    const bf16* kbase = k  + ((size_t)(b * NH + h) * SEQ) * HD;
    const bf16* vbase = vt + ((size_t)(b * NH + h) * HD) * SEQ;

    v8f o0 = { 0.f, 0.f, 0.f, 0.f, 0.f, 0.f, 0.f, 0.f };
    v8f o1 = { 0.f, 0.f, 0.f, 0.f, 0.f, 0.f, 0.f, 0.f };
    float m[8], l[8];
#pragma unroll
    for (int r = 0; r < 8; ++r) { m[r] = -1e30f; l[r] = 0.f; }

    for (int t0 = 0; t0 < SEQ; t0 += 32) {
        // Prefetch next K/V tiles (speculative global_prefetch_b8).
        __builtin_prefetch(kbase + (size_t)(t0 + 32 + n) * HD, 0, 1);
        __builtin_prefetch(vbase + (size_t)n * SEQ + t0 + 32, 0, 1);

        // K^T tiles as B operands (contiguous 32B per lane thanks to [s][hd] layout).
        const bf16* kp0 = kbase + (size_t)(t0 + n) * HD + hal * 16;
        const bf16* kp1 = kbase + (size_t)(t0 + 16 + n) * HD + hal * 16;
        v16bf kb0 = cat8(*(gv8bf*)kp0, *(gv8bf*)(kp0 + 8));
        v16bf kb1 = cat8(*(gv8bf*)kp1, *(gv8bf*)(kp1 + 8));

        v8f zc = { 0.f, 0.f, 0.f, 0.f, 0.f, 0.f, 0.f, 0.f };
        v8f sc0 = wmma_bf16(qa, kb0, zc);
        v8f sc1 = wmma_bf16(qa, kb1, zc);

        // Online softmax; C-layout row M = r + 8*hal spans the 16 lanes of a half-wave.
#pragma unroll
        for (int r = 0; r < 8; ++r) {
            float mx = fmaxf(sc0[r], sc1[r]);
            mx = fmaxf(mx, __shfl_xor(mx, 1, 16));
            mx = fmaxf(mx, __shfl_xor(mx, 2, 16));
            mx = fmaxf(mx, __shfl_xor(mx, 4, 16));
            mx = fmaxf(mx, __shfl_xor(mx, 8, 16));
            const float mnew = fmaxf(m[r], mx);
            const float corr = __expf(m[r] - mnew);
            m[r] = mnew;
            const float p0 = __expf(sc0[r] - mnew);
            const float p1 = __expf(sc1[r] - mnew);
            float rs = p0 + p1;
            rs += __shfl_xor(rs, 1, 16);
            rs += __shfl_xor(rs, 2, 16);
            rs += __shfl_xor(rs, 4, 16);
            rs += __shfl_xor(rs, 8, 16);
            l[r] = l[r] * corr + rs;
            o0[r] *= corr;
            o1[r] *= corr;
            const int M = r + hal * 8;
            P[w][M][n]      = (bf16)p0;          // C layout -> LDS
            P[w][M][16 + n] = (bf16)p1;
        }

        // Re-read P in A layout (same-wave LDS ops are in order; no barrier needed).
        const bf16* pp = &P[w][n][hal * 16];
        v16bf pa = cat8(*reinterpret_cast<const v8bf*>(pp),
                        *reinterpret_cast<const v8bf*>(pp + 8));

        // V B-operands from transposed V: contiguous 32B per lane.
        const bf16* vp0 = vbase + (size_t)n * SEQ + t0 + hal * 16;
        const bf16* vp1 = vbase + (size_t)(16 + n) * SEQ + t0 + hal * 16;
        v16bf vb0 = cat8(*(gv8bf*)vp0, *(gv8bf*)(vp0 + 8));
        v16bf vb1 = cat8(*(gv8bf*)vp1, *(gv8bf*)(vp1 + 8));

        o0 = wmma_bf16(pa, vb0, o0);
        o1 = wmma_bf16(pa, vb1, o1);
    }

    // Normalize and store transposed: aoT[b][s][c], c = h*32 + dhalf*16 + n.
#pragma unroll
    for (int r = 0; r < 8; ++r) {
        const float inv = 1.0f / l[r];
        const int s = q0 + r + hal * 8;
        mgbf16* dst = (mgbf16*)(aoT + ((size_t)b * SEQ + s) * CCH + h * HD);
        dst[n]      = (bf16)(o0[r] * inv);
        dst[16 + n] = (bf16)(o1[r] * inv);
    }
}

// ---------------------------------------------------------------------------
// Kernel 3: output projection.  out fp32 [B][C][S] = wo * aoT^T + bo.
// B-operand tile (8KB bf16) staged ONCE per workgroup via the Tensor Data
// Mover with LDS padding (row stride 512B+16B -> conflict-free b128 reads),
// instead of 8 waves re-reading it from global.
// ---------------------------------------------------------------------------
__global__ void __launch_bounds__(256) out_proj_kernel(
        const bf16* __restrict__ aoT,
        const float* __restrict__ wo, const float* __restrict__ bo,
        float* __restrict__ out) {
    __shared__ __align__(16) bf16 Bl[16][264];   // 16 rows x (256 data + 8 pad) bf16

    const int blk = blockIdx.x;
    const int b   = blk / (SEQ / 16);
    const int s0  = (blk % (SEQ / 16)) * 16;
    const int tid = threadIdx.x;

#if __has_builtin(__builtin_amdgcn_tensor_load_to_lds) && __has_builtin(__builtin_amdgcn_s_wait_tensorcnt)
    if (tid < 32) {  // wave 0 issues one TDM descriptor for the whole tile
        const unsigned lds_off =
            (unsigned)(uintptr_t)(__attribute__((address_space(3))) bf16*)&Bl[0][0];
        const unsigned long long ga =
            (unsigned long long)(uintptr_t)(aoT + ((size_t)b * SEQ + s0) * CCH);
        u32x4 g0;
        g0[0] = 1u;                                       // count=1 valid user descriptor
        g0[1] = lds_off;                                  // LDS byte address
        g0[2] = (unsigned)(ga & 0xFFFFFFFFu);             // global_addr[31:0]
        g0[3] = (unsigned)((ga >> 32) & 0x01FFFFFFu)      // global_addr[56:32]
              | (2u << 30);                               // type = 2 (image)
        i32x8 g1;
        g1[0] = (int)((1u << 16)       // data_size = 2 bytes
                    | (1u << 20)       // pad_enable
                    | (6u << 22)       // pad_interval: every 128 dwords (512B row)
                    | (3u << 25));     // pad_amount: 4 dwords (16B)
        g1[1] = (int)(256u << 16);     // tensor_dim0 = 256 elements
        g1[2] = (int)(16u << 16);      // tensor_dim1 = 16
        g1[3] = (int)(256u << 16);     // tile_dim0 = 256
        g1[4] = 16;                    // tile_dim1 = 16, tile_dim2 = 0
        g1[5] = 256;                   // tensor_dim0_stride = 256 elements
        g1[6] = 0;
        g1[7] = 0;
        i32x4 gz4 = { 0, 0, 0, 0 };        // groups 2/3 unused (2D tensor)
        i32x8 gz8 = { 0, 0, 0, 0, 0, 0, 0, 0 };
        // clang-23 / therock-10.0 form: 6 args (extra int32x8 group before cpol)
        __builtin_amdgcn_tensor_load_to_lds(g0, g1, gz4, gz4, gz8, 0);
        __builtin_amdgcn_s_wait_tensorcnt(0);
    }
    __syncthreads();
#else
    for (int id = tid; id < 512; id += 256) {   // fallback: plain cooperative copy
        const int row = id >> 5, seg = id & 31;
        *reinterpret_cast<v8bf*>(&Bl[row][seg * 8]) =
            *(gv8bf*)(aoT + ((size_t)b * SEQ + s0 + row) * CCH + seg * 8);
    }
    __syncthreads();
#endif

    const int w    = tid >> 5;
    const int lane = tid & 31;
    const int n    = lane & 15;
    const int hal  = lane >> 4;

    for (int mi = 0; mi < 2; ++mi) {
        const int cTile = (w * 2 + mi) * 16;
        v8f acc = { 0.f, 0.f, 0.f, 0.f, 0.f, 0.f, 0.f, 0.f };
#pragma unroll
        for (int kc = 0; kc < 8; ++kc) {
            const int c0 = kc * 32;
            v16bf a = load_w_tileA(wo + (size_t)(cTile + n) * CCH + c0 + hal * 8);
            const bf16* bp = &Bl[n][c0 + hal * 8];
            v16bf bb = cat8(*reinterpret_cast<const v8bf*>(bp),
                            *reinterpret_cast<const v8bf*>(bp + 16));
            acc = wmma_bf16(a, bb, acc);
        }
        const int s = s0 + n;
        const int cbase = cTile + hal * 8;
#pragma unroll
        for (int r = 0; r < 8; ++r) {
            const int c = cbase + r;
            ((mgfloat*)out)[((size_t)b * CCH + c) * SEQ + s] = acc[r] + ((gfloat*)bo)[c];
        }
    }
}

// ---------------------------------------------------------------------------
extern "C" void kernel_launch(void* const* d_in, const int* in_sizes, int n_in,
                              void* d_out, int out_size, void* d_ws, size_t ws_size,
                              hipStream_t stream) {
    const float* x  = (const float*)d_in[0];
    const float* wq = (const float*)d_in[1];
    const float* bq = (const float*)d_in[2];
    const float* wk = (const float*)d_in[3];
    const float* bk = (const float*)d_in[4];
    const float* wv = (const float*)d_in[5];
    const float* bv = (const float*)d_in[6];
    const float* wo = (const float*)d_in[7];
    const float* bo = (const float*)d_in[8];
    float* out = (float*)d_out;

    const size_t elems = (size_t)BATCH * CCH * SEQ;   // 2,097,152 per tensor
    bf16* qb  = (bf16*)d_ws;
    bf16* kb  = qb + elems;
    bf16* vT  = kb + elems;
    bf16* aoT = vT + elems;

    dim3 blkDim(256);
    qkv_proj_kernel<<<dim3(BATCH * (SEQ / 16)), blkDim, 0, stream>>>(
        x, wq, bq, wk, bk, wv, bv, qb, kb, vT);
    flash_attn_kernel<<<dim3(BATCH * NH * (SEQ / 128)), blkDim, 0, stream>>>(
        qb, kb, vT, aoT);
    out_proj_kernel<<<dim3(BATCH * (SEQ / 16)), blkDim, 0, stream>>>(
        aoT, wo, bo, out);
}
